// CustomMultiheadAttention_12395275616450
// MI455X (gfx1250) — matrix operand order repaired and checked by reading, not compiled, not run on hardware
//
#include <hip/hip_runtime.h>

// Problem constants (from reference): L, S, N, E, H, D
#define LL 2048
#define SS 2048
#define NN 2
#define EE 512
#define HH 8
#define DD 64

// LDS score-row stride in floats: 2048 + 4 pad so 16 rows map to distinct
// bank groups for the phase-3 b128 fragment reads (4*16 = 64 banks).
#define SC_STRIDE 2052

typedef __attribute__((ext_vector_type(16))) __bf16 v16bf;
typedef __attribute__((ext_vector_type(8)))  __bf16 v8bf;
typedef __attribute__((ext_vector_type(4)))  __bf16 v4bf;
typedef __attribute__((ext_vector_type(8)))  float  v8f;
typedef __attribute__((ext_vector_type(4)))  float  v4f;

static __device__ __forceinline__ v8f vzero8() {
  v8f z;
#pragma unroll
  for (int i = 0; i < 8; ++i) z[i] = 0.0f;
  return z;
}

// ---------------- WMMA fragment loaders (wave32, 16x16x32 bf16) ----------------
// A-matrix 16x32: lane = m + 16*kh; elems 0..7 = K[kh*8..+7], 8..15 = K[16+kh*8..+7]
// B-matrix 32x16: lane = n + 16*kh; elems 0..7 = K[kh*8..+7], 8..15 = K[16+kh*8..+7]
// Both read row-major sources where "row index" is (lane & 15).

static __device__ __forceinline__ v16bf a_frag_bf16(const __bf16* base, int stride, int k0) {
  int lane = threadIdx.x & 31;
  const __bf16* p = base + (size_t)(lane & 15) * stride + k0 + ((lane >> 4) << 3);
  v8bf lo = *(const v8bf*)p;
  v8bf hi = *(const v8bf*)(p + 16);
  v16bf f;
#pragma unroll
  for (int j = 0; j < 8; ++j) { f[j] = lo[j]; f[j + 8] = hi[j]; }
  return f;
}

static __device__ __forceinline__ v16bf a_frag_f32(const float* base, int stride, int k0) {
  int lane = threadIdx.x & 31;
  const float* p = base + (size_t)(lane & 15) * stride + k0 + ((lane >> 4) << 3);
  v4f x0 = *(const v4f*)p;
  v4f x1 = *(const v4f*)(p + 4);
  v4f y0 = *(const v4f*)(p + 16);
  v4f y1 = *(const v4f*)(p + 20);
  v16bf f;
#pragma unroll
  for (int j = 0; j < 4; ++j) {
    f[j]      = (__bf16)x0[j];
    f[j + 4]  = (__bf16)x1[j];
    f[j + 8]  = (__bf16)y0[j];
    f[j + 12] = (__bf16)y1[j];
  }
  return f;
}

// A fragment sourced from the LDS score block (f32, SC_STRIDE row pitch),
// scaled by the row's softmax 1/sum during bf16 conversion.
static __device__ __forceinline__ v16bf a_frag_lds(const float* base, int k0, float scale) {
  int lane = threadIdx.x & 31;
  const float* p = base + (lane & 15) * SC_STRIDE + k0 + ((lane >> 4) << 3);
  v16bf f;
#pragma unroll
  for (int j = 0; j < 8; ++j) {
    f[j]     = (__bf16)(p[j] * scale);
    f[j + 8] = (__bf16)(p[j + 16] * scale);
  }
  return f;
}

static __device__ __forceinline__ v16bf b_frag_bf16(const __bf16* base, int stride, int k0) {
  int lane = threadIdx.x & 31;
  const __bf16* p = base + (size_t)(lane & 15) * stride + k0 + ((lane >> 4) << 3);
  v8bf lo = *(const v8bf*)p;
  v8bf hi = *(const v8bf*)(p + 16);
  v16bf f;
#pragma unroll
  for (int j = 0; j < 8; ++j) { f[j] = lo[j]; f[j + 8] = hi[j]; }
  return f;
}

static __device__ __forceinline__ v16bf b_frag_f32(const float* base, int stride, int k0) {
  int lane = threadIdx.x & 31;
  const float* p = base + (size_t)(lane & 15) * stride + k0 + ((lane >> 4) << 3);
  v16bf f;
#pragma unroll
  for (int j = 0; j < 16; j += 4) {
    v4f x = *(const v4f*)(p + j + (j & 8));
    f[j]     = (__bf16)x[0];
    f[j + 1] = (__bf16)x[1];
    f[j + 2] = (__bf16)x[2];
    f[j + 3] = (__bf16)x[3];
  }
  return f;
}

static __device__ __forceinline__ v8f wmma_bf16(v16bf a, v16bf b, v8f c) {
  // (neg_a, A, neg_b, B, c_mod, C, reuse_a, reuse_b)
  return __builtin_amdgcn_wmma_f32_16x16x32_bf16(false, a, false, b, (short)0, c, false, false);
}

// ---------------- Kernel 1: QKV projections ----------------
// blockIdx.y: 0 = Q, 1 = K, 2 = V.  X (4096x512 f32) @ W^T + b.
// Each wave computes a 16x64 output block (4 accumulators): the A fragment is
// loaded once per k-step and reused for 4 back-to-back WMMAs.
// Q,K stored [n][h][row][d] bf16; V stored transposed [n][h][d][s] bf16.
__global__ void mha_qkv_proj(const float* __restrict__ q, const float* __restrict__ k,
                             const float* __restrict__ v,
                             const float* __restrict__ Wq, const float* __restrict__ bq,
                             const float* __restrict__ Wk, const float* __restrict__ bk,
                             const float* __restrict__ Wv, const float* __restrict__ bv,
                             __bf16* __restrict__ Qb, __bf16* __restrict__ Kb,
                             __bf16* __restrict__ Vt) {
  const int proj = blockIdx.y;
  const float* X    = (proj == 0) ? q  : (proj == 1) ? k  : v;
  const float* W    = (proj == 0) ? Wq : (proj == 1) ? Wk : Wv;
  const float* bias = (proj == 0) ? bq : (proj == 1) ? bk : bv;

  const int lane = threadIdx.x & 31;
  const int wid  = blockIdx.x * (blockDim.x >> 5) + (threadIdx.x >> 5);
  const int tm = wid >> 3;   // 0..255  (M = L*N = 4096 rows, 16 each)
  const int tb = wid & 7;    // 0..7    (E = 512 cols, 64 each)

  const float* Arow = X + (size_t)(tm * 16) * EE;

  v8f acc[4] = {vzero8(), vzero8(), vzero8(), vzero8()};
#pragma unroll 2
  for (int kt = 0; kt < EE; kt += 32) {
    const v16bf a = a_frag_f32(Arow, EE, kt);
#pragma unroll
    for (int c = 0; c < 4; ++c) {
      const float* Brow = W + (size_t)((tb * 4 + c) * 16) * EE;
      acc[c] = wmma_bf16(a, b_frag_f32(Brow, EE, kt), acc[c]);
    }
  }

  const int rbase = tm * 16 + ((lane >> 4) << 3);
#pragma unroll
  for (int c = 0; c < 4; ++c) {
    const int col = (tb * 4 + c) * 16 + (lane & 15);   // e
    const int h = col >> 6, d = col & 63;
    const float bval = bias[col];
#pragma unroll
    for (int j = 0; j < 8; ++j) {
      const int r = rbase + j;               // row in (seq*N) flattening
      const int sl = r >> 1;                 // l (or s for K/V), since N==2
      const int n  = r & 1;
      const float val = acc[c][j] + bval;
      if (proj == 0) {
        Qb[(((size_t)(n * HH + h)) * LL + sl) * DD + d] = (__bf16)val;
      } else if (proj == 1) {
        Kb[(((size_t)(n * HH + h)) * SS + sl) * DD + d] = (__bf16)val;
      } else {
        Vt[(((size_t)(n * HH + h)) * DD + d) * SS + sl] = (__bf16)val;
      }
    }
  }
}

// ---------------- Kernel 2: fused scores + softmax + PV ----------------
// One workgroup (8 waves) per (head-pair, 16-row L tile).
// LDS stages the 16 x 2048 f32 score block; bias is folded in during the
// coalesced softmax sweep; attn_weights stream out with nontemporal stores;
// PV partials reduce via ds_add_f32 atomics.
__global__ void __launch_bounds__(256)
mha_attention_fused(const __bf16* __restrict__ Qb, const __bf16* __restrict__ Kb,
                    const __bf16* __restrict__ Vt, const float* __restrict__ bias,
                    float* __restrict__ Wgt, __bf16* __restrict__ AO) {
  __shared__ __align__(16) float sc[16 * SC_STRIDE];   // 131,328 B score block
  __shared__ __align__(16) float accum[16 * DD];       // 4 KB PV accumulator
  __shared__ float redbuf[16 * 16];
  __shared__ float rowinv[16];

  const int t    = threadIdx.x;
  const int lane = t & 31;
  const int wave = t >> 5;
  const int pair = blockIdx.x;              // h*N + n  (fastest: L2 reuse of K/V)
  const int lt   = blockIdx.y;              // 16-row L tile
  const int h = pair >> 1, n = pair & 1;
  const int hd = n * HH + h;

  // zero PV accumulator up front (used after two barriers)
  {
    v4f z; z[0] = z[1] = z[2] = z[3] = 0.0f;
    *(v4f*)&accum[t * 4] = z;
  }

  const __bf16* Qh = Qb + (size_t)hd * LL * DD + (size_t)(lt * 16) * DD;
  const __bf16* Kh = Kb + (size_t)hd * SS * DD;
  const __bf16* Vh = Vt + (size_t)hd * DD * SS;

  // ---- Phase 1: raw scores (16 x S) * 1/sqrt(D) -> LDS ----
  const v16bf qa0 = a_frag_bf16(Qh, DD, 0);
  const v16bf qa1 = a_frag_bf16(Qh, DD, 32);
  const int l0 = (lane >> 4) << 3;
  for (int ts = 0; ts < 16; ++ts) {
    const int st = wave * 16 + ts;          // s-tile 0..127
    const __bf16* Kt = Kh + (size_t)(st * 16) * DD;
    if (ts < 15) __builtin_prefetch(Kt + 16 * DD, 0, 0);   // next K tile
    v8f acc = vzero8();
    acc = wmma_bf16(qa0, b_frag_bf16(Kt, DD, 0),  acc);
    acc = wmma_bf16(qa1, b_frag_bf16(Kt, DD, 32), acc);
    const int s = st * 16 + (lane & 15);
#pragma unroll
    for (int j = 0; j < 8; ++j) {
      sc[(l0 + j) * SC_STRIDE + s] = acc[j] * 0.125f;
    }
  }
  __syncthreads();

  // ---- Phase 2: bias add + softmax over S for 16 rows ----
  const int r  = t >> 4;         // row 0..15 (16 threads per row)
  const int c4 = (t & 15) * 4;   // float4 column phase
  float* srow = &sc[r * SC_STRIDE];
  const float* brow = bias + (size_t)(lt * 16 + r) * SS;   // coalesced v4 reads

  float m = -3.402823466e38f;
#pragma unroll 8
  for (int i = 0; i < 32; ++i) {
    v4f x = *(const v4f*)&srow[c4 + 64 * i];
    v4f b = *(const v4f*)&brow[c4 + 64 * i];
    x = x + b;
    *(v4f*)&srow[c4 + 64 * i] = x;
    m = fmaxf(m, fmaxf(fmaxf(x[0], x[1]), fmaxf(x[2], x[3])));
  }
  redbuf[r * 16 + (t & 15)] = m;
  __syncthreads();
#pragma unroll
  for (int j = 0; j < 16; ++j) m = fmaxf(m, redbuf[r * 16 + j]);
  __syncthreads();   // before redbuf reuse

  float ssum = 0.0f;
#pragma unroll 8
  for (int i = 0; i < 32; ++i) {
    v4f x = *(const v4f*)&srow[c4 + 64 * i];
    v4f e;
    e[0] = __expf(x[0] - m); e[1] = __expf(x[1] - m);
    e[2] = __expf(x[2] - m); e[3] = __expf(x[3] - m);
    *(v4f*)&srow[c4 + 64 * i] = e;           // keep unnormalized exp in LDS
    ssum += e[0] + e[1] + e[2] + e[3];
  }
  redbuf[r * 16 + (t & 15)] = ssum;
  __syncthreads();
  ssum = 0.0f;
#pragma unroll
  for (int j = 0; j < 16; ++j) ssum += redbuf[r * 16 + j];
  const float inv = 1.0f / ssum;
  if ((t & 15) == 0) rowinv[r] = inv;

  // stream normalized weights to global; never re-read -> nontemporal
  float* wrow = Wgt + (size_t)pair * LL * SS + (size_t)(lt * 16 + r) * SS;
#pragma unroll 8
  for (int i = 0; i < 32; ++i) {
    v4f e = *(const v4f*)&srow[c4 + 64 * i];
    v4f p;
    p[0] = e[0] * inv; p[1] = e[1] * inv; p[2] = e[2] * inv; p[3] = e[3] * inv;
    __builtin_nontemporal_store(p, (v4f*)&wrow[c4 + 64 * i]);
  }
  __syncthreads();

  // ---- Phase 3: PV. Waves partition S (256 cols each); 1/sum folded into A ----
  const float ascale = rowinv[lane & 15];
  const int kbase = wave * 256;
#pragma unroll
  for (int dt = 0; dt < 4; ++dt) {
    const __bf16* Vd = Vh + (size_t)(dt * 16) * SS;
    v8f acc = vzero8();
#pragma unroll
    for (int kk = 0; kk < 8; ++kk) {
      const int k0 = kbase + kk * 32;
      acc = wmma_bf16(a_frag_lds(sc, k0, ascale), b_frag_bf16(Vd, SS, k0), acc);
    }
#pragma unroll
    for (int j = 0; j < 8; ++j)
      atomicAdd(&accum[(l0 + j) * DD + dt * 16 + (lane & 15)], acc[j]);
  }
  __syncthreads();

  // ---- write AO tile (16 x 64) as bf16 (one 8B vector store per thread) ----
  {
    const int idx = t * 4;            // 0..1020
    const int lr  = idx >> 6;         // local row
    const int d0  = idx & 63;
    const int l   = lt * 16 + lr;
    v4bf o;
    o[0] = (__bf16)accum[idx];
    o[1] = (__bf16)accum[idx + 1];
    o[2] = (__bf16)accum[idx + 2];
    o[3] = (__bf16)accum[idx + 3];
    *(v4bf*)(AO + ((size_t)(l * NN + n)) * EE + h * DD + d0) = o;
  }
}

// ---------------- Kernel 3: out = AO @ Wo^T + bo (f32 output) ----------------
// Same 4-way register blocking as the projection kernel.
__global__ void mha_oproj(const __bf16* __restrict__ AO, const float* __restrict__ Wo,
                          const float* __restrict__ bo, float* __restrict__ out) {
  const int lane = threadIdx.x & 31;
  const int wid  = blockIdx.x * (blockDim.x >> 5) + (threadIdx.x >> 5);
  const int tm = wid >> 3;   // 0..255
  const int tb = wid & 7;    // 0..7

  const __bf16* Arow = AO + (size_t)(tm * 16) * EE;

  v8f acc[4] = {vzero8(), vzero8(), vzero8(), vzero8()};
#pragma unroll 2
  for (int kt = 0; kt < EE; kt += 32) {
    const v16bf a = a_frag_bf16(Arow, EE, kt);
#pragma unroll
    for (int c = 0; c < 4; ++c) {
      const float* Brow = Wo + (size_t)((tb * 4 + c) * 16) * EE;
      acc[c] = wmma_bf16(a, b_frag_f32(Brow, EE, kt), acc[c]);
    }
  }

  const int rbase = tm * 16 + ((lane >> 4) << 3);
#pragma unroll
  for (int c = 0; c < 4; ++c) {
    const int col = (tb * 4 + c) * 16 + (lane & 15);
    const float bval = bo[col];
#pragma unroll
    for (int j = 0; j < 8; ++j) {
      out[(size_t)(rbase + j) * EE + col] = acc[c][j] + bval;
    }
  }
}

// ---------------- Host launcher ----------------
extern "C" void kernel_launch(void* const* d_in, const int* in_sizes, int n_in,
                              void* d_out, int out_size, void* d_ws, size_t ws_size,
                              hipStream_t stream) {
  const float* q    = (const float*)d_in[0];
  const float* k    = (const float*)d_in[1];
  const float* v    = (const float*)d_in[2];
  const float* bias = (const float*)d_in[3];
  const float* Wq   = (const float*)d_in[4];
  const float* bq   = (const float*)d_in[5];
  const float* Wk   = (const float*)d_in[6];
  const float* bk   = (const float*)d_in[7];
  const float* Wv   = (const float*)d_in[8];
  const float* bv   = (const float*)d_in[9];
  const float* Wo   = (const float*)d_in[10];
  const float* bo   = (const float*)d_in[11];

  // Workspace (bf16): Qb, Kb (per-head row-major), Vt (per-head transposed), AO
  const size_t headElems = (size_t)NN * HH * LL * DD;   // 2,097,152
  __bf16* Qb = (__bf16*)d_ws;
  __bf16* Kb = Qb + headElems;
  __bf16* Vt = Kb + headElems;
  __bf16* AO = Vt + headElems;

  float* out = (float*)d_out;                   // attn_out: L*N*E floats
  float* Wgt = out + (size_t)LL * NN * EE;      // attn_weights: H*N*L*S floats

  // 1) QKV projections: 2048 waves per projection (16x64 block each), 8/block
  mha_qkv_proj<<<dim3(256, 3), 256, 0, stream>>>(q, k, v, Wq, bq, Wk, bk, Wv, bv,
                                                 Qb, Kb, Vt);
  // 2) fused attention: pair fastest so all heads' K/V + bias stay L2-resident
  mha_attention_fused<<<dim3(HH * NN, LL / 16), 256, 0, stream>>>(Qb, Kb, Vt, bias,
                                                                  Wgt, AO);
  // 3) output projection
  mha_oproj<<<dim3(256), 256, 0, stream>>>(AO, Wo, bo, out);
}
